// Linear_1726576855246
// MI455X (gfx1250) — matrix operand order repaired and checked by reading, not compiled, not run on hardware
//
#include <hip/hip_runtime.h>

typedef __attribute__((ext_vector_type(16))) _Float16 v16h;
typedef __attribute__((ext_vector_type(8)))  _Float16 v8h;
typedef __attribute__((ext_vector_type(4)))  _Float16 v4h;
typedef __attribute__((ext_vector_type(8)))  float    v8f;

#define TILE_M 128
#define TILE_N 128
#define K_STEP 64
#define LDS_K  72   // 64 + 8-half pad -> 144B row stride (36 banks), conflict-free b128 reads

// O[M,N] = X[M,K] * W[N,K]^T * scale[N]
// X fp32, W int32 (int8-valued), scale fp32. f16 WMMA with f32 accumulate.
__global__ __launch_bounds__(256, 1)
void int8w_linear_wmma(const float* __restrict__ X,
                       const int*   __restrict__ W,
                       const float* __restrict__ SC,
                       float* __restrict__ O,
                       int M, int N, int K)
{
    __shared__ __align__(16) _Float16 lsA[2][TILE_M][LDS_K];
    __shared__ __align__(16) _Float16 lsB[2][TILE_N][LDS_K];

    const int t    = threadIdx.x;
    const int lane = t & 31;
    const int wave = t >> 5;
    const int wr   = wave & 3;     // 4 row groups  (32 rows each)
    const int wc   = wave >> 2;    // 2 col groups  (64 cols each)
    const int l16  = lane & 15;
    const int h    = lane >> 4;    // half-wave select per ISA fragment layouts

    const int mTile = blockIdx.y * TILE_M;
    const int nTile = blockIdx.x * TILE_N;

    // Global-load mapping: 256 threads x 8 iters x float4 = 128x64 tile
    const int lrow = t >> 4;         // 0..15
    const int lcol = (t & 15) * 4;   // 0..60 step 4

    v8f zero = {};
    v8f acc[2][4];
    #pragma unroll
    for (int i = 0; i < 2; ++i)
        #pragma unroll
        for (int j = 0; j < 4; ++j) acc[i][j] = zero;

    const int stages = K / K_STEP;

    float4 ar[8];
    int4   br[8];

    // ---- prologue: stage 0 -> regs -> convert -> LDS buf 0 ----
    #pragma unroll
    for (int it = 0; it < 8; ++it) {
        const int row = it * 16 + lrow;
        ar[it] = *(const float4*)(X + (size_t)(mTile + row) * K + lcol);
        br[it] = *(const int4  *)(W + (size_t)(nTile + row) * K + lcol);
    }
    #pragma unroll
    for (int it = 0; it < 8; ++it) {
        const int row = it * 16 + lrow;
        v4h va; va.x = (_Float16)ar[it].x; va.y = (_Float16)ar[it].y;
                va.z = (_Float16)ar[it].z; va.w = (_Float16)ar[it].w;
        *(v4h*)&lsA[0][row][lcol] = va;
        v4h vb; vb.x = (_Float16)br[it].x; vb.y = (_Float16)br[it].y;
                vb.z = (_Float16)br[it].z; vb.w = (_Float16)br[it].w;
        *(v4h*)&lsB[0][row][lcol] = vb;
    }
    __syncthreads();

    for (int s = 0; s < stages; ++s) {
        const int cur = s & 1;

        // prefetch next stage into registers while WMMAs run
        if (s + 1 < stages) {
            const int k0n = (s + 1) * K_STEP;
            #pragma unroll
            for (int it = 0; it < 8; ++it) {
                const int row = it * 16 + lrow;
                ar[it] = *(const float4*)(X + (size_t)(mTile + row) * K + k0n + lcol);
                br[it] = *(const int4  *)(W + (size_t)(nTile + row) * K + k0n + lcol);
            }
        }

        // ---- compute: two k=32 sub-steps, 8 WMMAs each ----
        #pragma unroll
        for (int ks = 0; ks < 2; ++ks) {
            const int kb = ks * 32;
            v16h Af[2], Bf[4];

            // A 16x32 f16 fragment: lane holds row M=l16, K = h*16 .. h*16+15 contiguous
            #pragma unroll
            for (int mf = 0; mf < 2; ++mf) {
                const _Float16* p = &lsA[cur][wr * 32 + mf * 16 + l16][kb + h * 16];
                v8h lo = *(const v8h*)p;
                v8h hi = *(const v8h*)(p + 8);
                Af[mf] = __builtin_shufflevector(lo, hi, 0,1,2,3,4,5,6,7,8,9,10,11,12,13,14,15);
            }
            // B 32x16 f16 fragment: lane holds col N=l16, K = h*16 .. h*16+15 contiguous
            #pragma unroll
            for (int nf = 0; nf < 4; ++nf) {
                const _Float16* p = &lsB[cur][wc * 64 + nf * 16 + l16][kb + h * 16];
                v8h lo = *(const v8h*)p;
                v8h hi = *(const v8h*)(p + 8);
                Bf[nf] = __builtin_shufflevector(lo, hi, 0,1,2,3,4,5,6,7,8,9,10,11,12,13,14,15);
            }

            #pragma unroll
            for (int mf = 0; mf < 2; ++mf)
                #pragma unroll
                for (int nf = 0; nf < 4; ++nf)
                    acc[mf][nf] = __builtin_amdgcn_wmma_f32_16x16x32_f16(
                        false, Af[mf], false, Bf[nf],
                        (short)0, acc[mf][nf], false, false);
        }

        // ---- convert prefetched regs into the other LDS buffer ----
        if (s + 1 < stages) {
            const int nxt = cur ^ 1;
            #pragma unroll
            for (int it = 0; it < 8; ++it) {
                const int row = it * 16 + lrow;
                v4h va; va.x = (_Float16)ar[it].x; va.y = (_Float16)ar[it].y;
                        va.z = (_Float16)ar[it].z; va.w = (_Float16)ar[it].w;
                *(v4h*)&lsA[nxt][row][lcol] = va;
                v4h vb; vb.x = (_Float16)br[it].x; vb.y = (_Float16)br[it].y;
                        vb.z = (_Float16)br[it].z; vb.w = (_Float16)br[it].w;
                *(v4h*)&lsB[nxt][row][lcol] = vb;
            }
        }
        __syncthreads();
    }

    // ---- epilogue: per-channel scale, store ----
    // C/D layout: lane holds col N = l16; VGPR v holds row M = v + 8*h
    #pragma unroll
    for (int nf = 0; nf < 4; ++nf) {
        const int n = nTile + wc * 64 + nf * 16 + l16;
        const float sc = SC[n];
        #pragma unroll
        for (int mf = 0; mf < 2; ++mf) {
            const int mBase = mTile + wr * 32 + mf * 16 + h * 8;
            #pragma unroll
            for (int v = 0; v < 8; ++v) {
                O[(size_t)(mBase + v) * N + n] = acc[mf][nf][v] * sc;
            }
        }
    }
}

extern "C" void kernel_launch(void* const* d_in, const int* in_sizes, int n_in,
                              void* d_out, int out_size, void* d_ws, size_t ws_size,
                              hipStream_t stream) {
    const float* x  = (const float*)d_in[0];
    const int*   w  = (const int*)  d_in[1];
    const float* sc = (const float*)d_in[2];
    float*       o  = (float*)d_out;

    const int M = 4 * 2048;   // B*S
    const int N = 11008;      // OUT
    const int K = 4096;       // IN

    dim3 grid(N / TILE_N, M / TILE_M);   // 86 x 64
    int8w_linear_wmma<<<grid, 256, 0, stream>>>(x, w, sc, o, M, N, K);
}
